// PerformerWrapper_74002286510202
// MI455X (gfx1250) — matrix operand order repaired and checked by reading, not compiled, not run on hardware
//
#include <hip/hip_runtime.h>

// Performer / FAVOR+ attention for MI455X (gfx1250), wave32 + WMMA bf16.
// All GEMM-shaped work runs through v_wmma_f32_16x16x32_bf16 (fp32 accum).
// K-loops are double-buffered in LDS with register-staged global loads so
// the memory pipe stays ahead of the matrix pipe.

#define DEVI __device__ __forceinline__

constexpr int BB  = 16;    // batch
constexpr int CCH = 512;   // channels
constexpr int NN  = 1024;  // tokens (H*W)
constexpr int HH  = 8;     // heads
constexpr int DH  = 64;    // head dim
constexpr int MMF = 256;   // random features m
constexpr int BH  = BB * HH;  // 128
constexpr int BN  = BB * NN;  // 16384

typedef __attribute__((ext_vector_type(16))) __bf16 v16bf;
typedef __attribute__((ext_vector_type(8)))  float  v8f;

union FragU { v16bf v; unsigned u[8]; };

DEVI unsigned short f2bf(float f) {
  unsigned u = __float_as_uint(f);
  u += 0x7fffu + ((u >> 16) & 1u);   // round-to-nearest-even
  return (unsigned short)(u >> 16);
}
DEVI float bf2f(unsigned short h) { return __uint_as_float(((unsigned)h) << 16); }
DEVI unsigned pack2(float lo, float hi) {
  return (unsigned)f2bf(lo) | ((unsigned)f2bf(hi) << 16);
}

DEVI v8f wmma_bf16(v16bf a, v16bf b, v8f c) {
  return __builtin_amdgcn_wmma_f32_16x16x32_bf16(false, a, false, b, (short)0, c,
                                                 false, false);
}

// A fragment 16x32 bf16: lanes 0-15 rows M, K-pair map per CDNA5 ISA 7.12.2.
// LDS layout: row-major [M][k], leading dim ld (halfwords, even).
DEVI v16bf load_frag_a(const unsigned short* base, int ld) {
  const int lane = threadIdx.x & 31;
  const int M = lane & 15, half = lane >> 4;
  FragU f;
  const unsigned short* row = base + M * ld;
#pragma unroll
  for (int v = 0; v < 8; ++v) {
    int k = ((v & 3) << 1) + (half << 3) + ((v >> 2) << 4);
    f.u[v] = *(const unsigned*)(row + k);
  }
  return f.v;
}
// B fragment 32x16 (KxN) bf16: lane column N, VGPR r -> K = 2(r&3) + half*8 + (r>>2)*16.
// LDS layout: N-major [n][k] (K-major rows), leading dim ld.
DEVI v16bf load_frag_b(const unsigned short* base, int ld) {
  const int lane = threadIdx.x & 31;
  const int Ncol = lane & 15, half = lane >> 4;
  FragU f;
  const unsigned short* row = base + Ncol * ld;
#pragma unroll
  for (int r = 0; r < 8; ++r) {
    int k = ((r & 3) << 1) + (half << 3) + ((r >> 2) << 4);
    f.u[r] = *(const unsigned*)(row + k);
  }
  return f.v;
}

DEVI void atomicMaxFloat(float* addr, float val) {
  if (val >= 0.f) atomicMax((int*)addr, __float_as_int(val));
  else            atomicMin((unsigned int*)addr, __float_as_uint(val));
}

// ---------------------------------------------------------------- init ----
__global__ void perf_init_kernel(float* ksum, float* keymax, float* denom) {
  int i = blockIdx.x * blockDim.x + threadIdx.x;
  if (i < BH * MMF) ksum[i] = 0.f;
  if (i < BH)       keymax[i] = -3.0e38f;
  if (i < BH * NN)  denom[i] = 0.f;
}

// ------------------------------------------------------ QKV projection ----
// t[b,n,c] = x[b,c,n];  Q/K/V = t @ W{q,k,v}  (bf16 out, [BN][C])
__global__ __launch_bounds__(256)
void perf_qkv_kernel(const float* __restrict__ x,
                     const float* __restrict__ Wq, const float* __restrict__ Wk,
                     const float* __restrict__ Wv,
                     unsigned short* __restrict__ Q, unsigned short* __restrict__ Kq,
                     unsigned short* __restrict__ V) {
  __shared__ alignas(16) unsigned short As[2][64 * 34];
  __shared__ alignas(16) unsigned short Bs[2][64 * 34];
  const int tid = threadIdx.x, lane = tid & 31, wave = tid >> 5;
  const int row0 = blockIdx.x * 64, col0 = blockIdx.y * 64;
  const int b = row0 >> 10, n0 = row0 & 1023;
  const float* W = (blockIdx.z == 0) ? Wq : (blockIdx.z == 1) ? Wk : Wv;
  unsigned short* Out = (blockIdx.z == 0) ? Q : (blockIdx.z == 1) ? Kq : V;
  const int wr = wave >> 2, wc = wave & 3;     // tiles (wr,wc) and (wr+2,wc)

  const int ar = tid & 63, ak = tid >> 6;      // row 0..63, k 0..3 (+4*i)
  float aR[8], bR[8];
  auto gload = [&](int k0) {
#pragma unroll
    for (int i = 0; i < 8; ++i) {
      aR[i] = x[(size_t)b * CCH * NN + (size_t)(k0 + ak + 4 * i) * NN + n0 + ar];
      bR[i] = W[(size_t)(k0 + ak + 4 * i) * CCH + col0 + ar];
    }
  };
  auto stage = [&](int buf) {
#pragma unroll
    for (int i = 0; i < 8; ++i) {
      As[buf][ar * 34 + ak + 4 * i] = f2bf(aR[i]);
      Bs[buf][ar * 34 + ak + 4 * i] = f2bf(bR[i]);
    }
  };

  gload(0);
  stage(0);
  __syncthreads();
  v8f acc0 = {}, acc1 = {};
  for (int t = 0; t < CCH / 32; ++t) {
    if (t < CCH / 32 - 1) {
      gload((t + 1) * 32);
      if (t < CCH / 32 - 2)   // one speculative prefetch per thread, uniform branch
        __builtin_prefetch(
            &x[(size_t)b * CCH * NN + (size_t)((t + 2) * 32 + ak) * NN + n0 + ar], 0, 1);
    }
    const unsigned short* Ab = As[t & 1];
    const unsigned short* Bb = Bs[t & 1];
    v16bf bf = load_frag_b(&Bb[wc * 16 * 34], 34);
    v16bf a0 = load_frag_a(&Ab[wr * 16 * 34], 34);
    v16bf a1 = load_frag_a(&Ab[(wr + 2) * 16 * 34], 34);
    acc0 = wmma_bf16(a0, bf, acc0);
    acc1 = wmma_bf16(a1, bf, acc1);
    if (t < CCH / 32 - 1) stage((t + 1) & 1);
    __syncthreads();
  }
  const int lcol = lane & 15, half = lane >> 4;
#pragma unroll
  for (int r = 0; r < 8; ++r) {
    int m0 = wr * 16 + r + half * 8;
    int m1 = (wr + 2) * 16 + r + half * 8;
    int gcol = col0 + wc * 16 + lcol;
    Out[(size_t)(row0 + m0) * CCH + gcol] = f2bf(acc0[r]);
    Out[(size_t)(row0 + m1) * CCH + gcol] = f2bf(acc1[r]);
  }
}

// ---------------------------------------------------- FAVOR+ features ----
// FMODE 0: reduce global max of data_dash (key stab)  -> keymax[bh]
// FMODE 1: kp = ratio*(exp(dd - diag - keymax)+eps), accumulate k_sum
// FMODE 2: qp = ratio*(exp(dd - diag - rowmax)+eps), denom = qp . k_sum
template <int FMODE>
__global__ __launch_bounds__(256)
void perf_feature_kernel(const unsigned short* __restrict__ data,
                         const float* __restrict__ proj,
                         unsigned short* __restrict__ outP,
                         float* __restrict__ ksum,
                         float* __restrict__ keymax,
                         float* __restrict__ denom) {
  __shared__ alignas(16) unsigned short As[64 * 66];
  __shared__ alignas(16) unsigned short Bs[256 * 66];
  __shared__ float diagLds[64];
  __shared__ float redLds[64 * 4];
  __shared__ float rowmaxLds[64 * 2];
  __shared__ float colLds[256];
  __shared__ float dnmLds[64];
  __shared__ float wmaxLds[8];
  const int tid = threadIdx.x, lane = tid & 31, wave = tid >> 5;
  const int lcol = lane & 15, half = lane >> 4;
  const int bh = blockIdx.y, b = bh >> 3, hd = bh & 7;
  const int n0 = blockIdx.x * 64;
  constexpr float kNorm = 0.35355339059327373f;  // 64^-0.25
  constexpr float kRatio = 0.0625f;              // 256^-0.5
  constexpr float kEps = 1e-4f;

  if (FMODE == 1) colLds[tid] = 0.f;
  if (FMODE == 2) {
    colLds[tid] = ksum[bh * MMF + tid];
    if (tid < 64) dnmLds[tid] = 0.f;
  }
  // A: (data * normalizer) rows [n0..n0+63], cols 0..63 of this head.
  // Register-staged: 8 pair loads in flight, then convert+store.
  {
    unsigned aR[8];
#pragma unroll
    for (int i = 0; i < 8; ++i) {
      int pidx = tid + 256 * i;
      int r = pidx >> 5, kp2 = (pidx & 31) << 1;
      aR[i] = *(const unsigned*)&data[(size_t)(b * NN + n0 + r) * CCH + hd * DH + kp2];
    }
#pragma unroll
    for (int i = 0; i < 8; ++i) {
      int pidx = tid + 256 * i;
      int r = pidx >> 5, kp2 = (pidx & 31) << 1;
      float lo = bf2f((unsigned short)aR[i]) * kNorm;
      float hi = bf2f((unsigned short)(aR[i] >> 16)) * kNorm;
      *(unsigned*)&As[r * 66 + kp2] = pack2(lo, hi);
    }
  }
  // B: proj [256][64] -> K-major rows; float2 (b64) loads, 4 chunks of 8.
#pragma unroll
  for (int c = 0; c < 4; ++c) {
    float2 pR[8];
#pragma unroll
    for (int i = 0; i < 8; ++i) {
      int pidx = tid + 256 * (c * 8 + i);
      int r = pidx >> 5, kp2 = (pidx & 31) << 1;
      pR[i] = *(const float2*)&proj[r * DH + kp2];
    }
#pragma unroll
    for (int i = 0; i < 8; ++i) {
      int pidx = tid + 256 * (c * 8 + i);
      int r = pidx >> 5, kp2 = (pidx & 31) << 1;
      *(unsigned*)&Bs[r * 66 + kp2] = pack2(pR[i].x, pR[i].y);
    }
  }
  __syncthreads();

  // diag = 0.5 * ||row||^2 of scaled data
  {
    int r = tid & 63, qq = tid >> 6;
    float s = 0.f;
#pragma unroll
    for (int k = 0; k < 16; ++k) {
      float v = bf2f(As[r * 66 + qq * 16 + k]);
      s += v * v;
    }
    redLds[r * 4 + qq] = s;
  }
  __syncthreads();
  if (tid < 64)
    diagLds[tid] = 0.5f * (redLds[tid * 4] + redLds[tid * 4 + 1] +
                           redLds[tid * 4 + 2] + redLds[tid * 4 + 3]);

  // wave -> row tile rt (0..3), col group cg (0..1) covering 8 col tiles
  const int rt = wave & 3, cg = wave >> 2;
  v16bf a0 = load_frag_a(&As[rt * 16 * 66], 66);
  v16bf a1 = load_frag_a(&As[rt * 16 * 66 + 32], 66);
  v8f acc[8];
#pragma unroll
  for (int c = 0; c < 8; ++c) {
    int ct = cg * 8 + c;
    v16bf b0 = load_frag_b(&Bs[ct * 16 * 66], 66);
    v16bf b1 = load_frag_b(&Bs[ct * 16 * 66 + 32], 66);
    v8f z = {};
    z = wmma_bf16(a0, b0, z);
    z = wmma_bf16(a1, b1, z);
    acc[c] = z;
  }
  __syncthreads();  // diagLds ready, As free

  if constexpr (FMODE == 0) {
    float mx = -3.0e38f;
#pragma unroll
    for (int c = 0; c < 8; ++c)
#pragma unroll
      for (int r = 0; r < 8; ++r) mx = fmaxf(mx, acc[c][r]);
#pragma unroll
    for (int m = 1; m <= 16; m <<= 1) mx = fmaxf(mx, __shfl_xor(mx, m, 32));
    if (lane == 0) wmaxLds[wave] = mx;
    __syncthreads();
    if (tid == 0) {
      float g = wmaxLds[0];
#pragma unroll
      for (int w = 1; w < 8; ++w) g = fmaxf(g, wmaxLds[w]);
      atomicMaxFloat(&keymax[bh], g);
    }
    return;
  }

  float stab[8];
  if constexpr (FMODE == 2) {
    float rmax[8];
#pragma unroll
    for (int r = 0; r < 8; ++r) rmax[r] = acc[0][r];
#pragma unroll
    for (int c = 1; c < 8; ++c)
#pragma unroll
      for (int r = 0; r < 8; ++r) rmax[r] = fmaxf(rmax[r], acc[c][r]);
#pragma unroll
    for (int m = 1; m <= 8; m <<= 1)
#pragma unroll
      for (int r = 0; r < 8; ++r) rmax[r] = fmaxf(rmax[r], __shfl_xor(rmax[r], m, 32));
    if (lcol == 0)
#pragma unroll
      for (int r = 0; r < 8; ++r)
        rowmaxLds[(rt * 16 + r + half * 8) * 2 + cg] = rmax[r];
    __syncthreads();
#pragma unroll
    for (int r = 0; r < 8; ++r) {
      int row = rt * 16 + r + half * 8;
      stab[r] = fmaxf(rowmaxLds[row * 2], rowmaxLds[row * 2 + 1]);
    }
  } else {
    float s = keymax[bh];
#pragma unroll
    for (int r = 0; r < 8; ++r) stab[r] = s;
  }

  float dg[8];
#pragma unroll
  for (int r = 0; r < 8; ++r) dg[r] = diagLds[rt * 16 + r + half * 8];

  float dnm[8];
  if (FMODE == 2)
#pragma unroll
    for (int r = 0; r < 8; ++r) dnm[r] = 0.f;

#pragma unroll
  for (int c = 0; c < 8; ++c) {
    int col = (cg * 8 + c) * 16 + lcol;
    float cs = 0.f;
#pragma unroll
    for (int r = 0; r < 8; ++r) {
      float v = kRatio * (__expf(acc[c][r] - dg[r] - stab[r]) + kEps);
      int row = rt * 16 + r + half * 8;
      outP[(size_t)(bh * NN + n0 + row) * MMF + col] = f2bf(v);
      if (FMODE == 1) cs += v;
      if (FMODE == 2) dnm[r] += v * colLds[col];
    }
    if constexpr (FMODE == 1) {
      cs += __shfl_xor(cs, 16, 32);
      if (half == 0) atomicAdd(&colLds[col], cs);  // ds_add_f32
    }
  }
  if constexpr (FMODE == 2) {
#pragma unroll
    for (int m = 1; m <= 8; m <<= 1)
#pragma unroll
      for (int r = 0; r < 8; ++r) dnm[r] += __shfl_xor(dnm[r], m, 32);
    if (lcol == 0)
#pragma unroll
      for (int r = 0; r < 8; ++r)
        atomicAdd(&dnmLds[rt * 16 + r + half * 8], dnm[r]);
  }
  __syncthreads();
  if constexpr (FMODE == 1) atomicAdd(&ksum[bh * MMF + tid], colLds[tid]);
  if constexpr (FMODE == 2)
    if (tid < 64) denom[bh * NN + n0 + tid] = dnmLds[tid];
}

// -------------------------------------------------- context = kp^T @ v ----
__global__ __launch_bounds__(256)
void perf_context_kernel(const unsigned short* __restrict__ kp,
                         const unsigned short* __restrict__ V,
                         unsigned short* __restrict__ ctx /* [bh][d][m] K-major */) {
  __shared__ alignas(16) unsigned short As[2][256 * 34];  // kp^T tile [j][nn]
  __shared__ alignas(16) unsigned short Bs[2][64 * 34];   // v tile   [d][nn]
  const int tid = threadIdx.x, lane = tid & 31, wave = tid >> 5;
  const int bh = blockIdx.x, b = bh >> 3, hd = bh & 7;
  const int jt0 = wave * 2;  // wave owns j-tiles jt0, jt0+1, all 4 d-tiles

  unsigned kR[16], vR[4];
  auto gload = [&](int n0) {
#pragma unroll
    for (int i = 0; i < 16; ++i) {  // 4096 kp pairs
      int pidx = tid + 256 * i;
      int nn = pidx >> 7, jp = (pidx & 127) << 1;
      kR[i] = *(const unsigned*)&kp[(size_t)(bh * NN + n0 + nn) * MMF + jp];
    }
#pragma unroll
    for (int i = 0; i < 4; ++i) {   // 1024 v pairs
      int pidx = tid + 256 * i;
      int nn = pidx >> 5, dp = (pidx & 31) << 1;
      vR[i] = *(const unsigned*)&V[(size_t)(b * NN + n0 + nn) * CCH + hd * DH + dp];
    }
  };
  auto stage = [&](int buf) {
#pragma unroll
    for (int i = 0; i < 16; ++i) {  // transpose into LDS
      int pidx = tid + 256 * i;
      int nn = pidx >> 7, jp = (pidx & 127) << 1;
      As[buf][jp * 34 + nn] = (unsigned short)kR[i];
      As[buf][(jp + 1) * 34 + nn] = (unsigned short)(kR[i] >> 16);
    }
#pragma unroll
    for (int i = 0; i < 4; ++i) {
      int pidx = tid + 256 * i;
      int nn = pidx >> 5, dp = (pidx & 31) << 1;
      Bs[buf][dp * 34 + nn] = (unsigned short)vR[i];
      Bs[buf][(dp + 1) * 34 + nn] = (unsigned short)(vR[i] >> 16);
    }
  };

  gload(0);
  stage(0);
  __syncthreads();
  v8f acc[2][4] = {};
  for (int t = 0; t < NN / 32; ++t) {
    if (t < NN / 32 - 1) gload((t + 1) * 32);
    const unsigned short* Ab = As[t & 1];
    const unsigned short* Bb = Bs[t & 1];
    v16bf af0 = load_frag_a(&Ab[jt0 * 16 * 34], 34);
    v16bf af1 = load_frag_a(&Ab[(jt0 + 1) * 16 * 34], 34);
#pragma unroll
    for (int c = 0; c < 4; ++c) {
      v16bf bf = load_frag_b(&Bb[c * 16 * 34], 34);
      acc[0][c] = wmma_bf16(af0, bf, acc[0][c]);
      acc[1][c] = wmma_bf16(af1, bf, acc[1][c]);
    }
    if (t < NN / 32 - 1) stage((t + 1) & 1);
    __syncthreads();
  }
  const int lcol = lane & 15, half = lane >> 4;
#pragma unroll
  for (int t = 0; t < 2; ++t)
#pragma unroll
    for (int c = 0; c < 4; ++c)
#pragma unroll
      for (int r = 0; r < 8; ++r) {
        int j = (jt0 + t) * 16 + r + half * 8;
        int d = c * 16 + lcol;
        ctx[(size_t)bh * DH * MMF + d * MMF + j] = f2bf(acc[t][c][r]);
      }
}

// ----------------------------------------- out = (qp @ context)/denom ----
__global__ __launch_bounds__(256)
void perf_attnout_kernel(const unsigned short* __restrict__ qp,
                         const unsigned short* __restrict__ ctx,
                         const float* __restrict__ denom,
                         float* __restrict__ O /* [BN][C] */) {
  __shared__ alignas(16) unsigned short As[2][64 * 34];
  __shared__ alignas(16) unsigned short Bs[2][64 * 34];
  __shared__ float dnmLds[64];
  const int tid = threadIdx.x, lane = tid & 31, wave = tid >> 5;
  const int bh = blockIdx.y, b = bh >> 3, hd = bh & 7;
  const int n0 = blockIdx.x * 64;
  if (tid < 64) dnmLds[tid] = denom[bh * NN + n0 + tid];
  const int wr = wave >> 2, wc = wave & 3;

  const int sr = tid >> 4;          // +16*i
  const int kp2 = (tid & 15) * 2;
  unsigned aR[4], bR[4];
  auto gload = [&](int k0) {
#pragma unroll
    for (int i = 0; i < 4; ++i) {
      int r = sr + 16 * i;
      aR[i] = *(const unsigned*)&qp[(size_t)(bh * NN + n0 + r) * MMF + k0 + kp2];
      bR[i] = *(const unsigned*)&ctx[(size_t)bh * DH * MMF + r * MMF + k0 + kp2];
    }
  };
  auto stage = [&](int buf) {
#pragma unroll
    for (int i = 0; i < 4; ++i) {
      int r = sr + 16 * i;
      *(unsigned*)&As[buf][r * 34 + kp2] = aR[i];
      *(unsigned*)&Bs[buf][r * 34 + kp2] = bR[i];
    }
  };

  gload(0);
  stage(0);
  __syncthreads();
  v8f acc0 = {}, acc1 = {};
  for (int t = 0; t < MMF / 32; ++t) {
    if (t < MMF / 32 - 1) gload((t + 1) * 32);
    const unsigned short* Ab = As[t & 1];
    const unsigned short* Bb = Bs[t & 1];
    v16bf bf = load_frag_b(&Bb[wc * 16 * 34], 34);
    v16bf a0 = load_frag_a(&Ab[wr * 16 * 34], 34);
    v16bf a1 = load_frag_a(&Ab[(wr + 2) * 16 * 34], 34);
    acc0 = wmma_bf16(a0, bf, acc0);
    acc1 = wmma_bf16(a1, bf, acc1);
    if (t < MMF / 32 - 1) stage((t + 1) & 1);
    __syncthreads();
  }
  const int lcol = lane & 15, half = lane >> 4;
#pragma unroll
  for (int r = 0; r < 8; ++r) {
    int m0 = wr * 16 + r + half * 8;
    int m1 = (wr + 2) * 16 + r + half * 8;
    int col = hd * DH + wc * 16 + lcol;
    O[(size_t)(b * NN + n0 + m0) * CCH + col] = acc0[r] / dnmLds[m0];
    O[(size_t)(b * NN + n0 + m1) * CCH + col] = acc1[r] / dnmLds[m1];
  }
}

// --------------------------- final projection, store as [B,C,H,W] ----
__global__ __launch_bounds__(256)
void perf_outproj_kernel(const float* __restrict__ Oin,
                         const float* __restrict__ Wo,
                         float* __restrict__ out) {
  __shared__ alignas(16) unsigned short As[2][64 * 34];
  __shared__ alignas(16) unsigned short Bs[2][64 * 34];
  const int tid = threadIdx.x, lane = tid & 31, wave = tid >> 5;
  const int row0 = blockIdx.x * 64, col0 = blockIdx.y * 64;
  const int b = row0 >> 10, n0 = row0 & 1023;
  const int wr = wave >> 2, wc = wave & 3;

  const int sr = tid >> 4, kp2 = (tid & 15) * 2;  // A staging: float2 loads
  const int bc = tid & 63, bk = tid >> 6;         // B staging
  float2 aR[4];
  float bR[8];
  auto gload = [&](int k0) {
#pragma unroll
    for (int i = 0; i < 4; ++i)
      aR[i] = *(const float2*)&Oin[(size_t)(row0 + sr + 16 * i) * CCH + k0 + kp2];
#pragma unroll
    for (int i = 0; i < 8; ++i)
      bR[i] = Wo[(size_t)(k0 + bk + 4 * i) * CCH + col0 + bc];
  };
  auto stage = [&](int buf) {
#pragma unroll
    for (int i = 0; i < 4; ++i)
      *(unsigned*)&As[buf][(sr + 16 * i) * 34 + kp2] = pack2(aR[i].x, aR[i].y);
#pragma unroll
    for (int i = 0; i < 8; ++i)
      Bs[buf][bc * 34 + bk + 4 * i] = f2bf(bR[i]);
  };

  gload(0);
  stage(0);
  __syncthreads();
  v8f acc0 = {}, acc1 = {};
  for (int t = 0; t < CCH / 32; ++t) {
    if (t < CCH / 32 - 1) gload((t + 1) * 32);
    const unsigned short* Ab = As[t & 1];
    const unsigned short* Bb = Bs[t & 1];
    v16bf bf = load_frag_b(&Bb[wc * 16 * 34], 34);
    v16bf a0 = load_frag_a(&Ab[wr * 16 * 34], 34);
    v16bf a1 = load_frag_a(&Ab[(wr + 2) * 16 * 34], 34);
    acc0 = wmma_bf16(a0, bf, acc0);
    acc1 = wmma_bf16(a1, bf, acc1);
    if (t < CCH / 32 - 1) stage((t + 1) & 1);
    __syncthreads();
  }
  const int lcol = lane & 15, half = lane >> 4;
#pragma unroll
  for (int r = 0; r < 8; ++r) {
    int m0 = wr * 16 + r + half * 8;
    int m1 = (wr + 2) * 16 + r + half * 8;
    int gcol = col0 + wc * 16 + lcol;
    out[(size_t)b * CCH * NN + (size_t)gcol * NN + (n0 + m0)] = acc0[r];
    out[(size_t)b * CCH * NN + (size_t)gcol * NN + (n0 + m1)] = acc1[r];
  }
}

// ------------------------------------------------------------ launch ----
extern "C" void kernel_launch(void* const* d_in, const int* in_sizes, int n_in,
                              void* d_out, int out_size, void* d_ws, size_t ws_size,
                              hipStream_t stream) {
  (void)in_sizes; (void)n_in; (void)out_size; (void)ws_size;
  const float* x    = (const float*)d_in[0];
  const float* Wq   = (const float*)d_in[1];
  const float* Wk   = (const float*)d_in[2];
  const float* Wv   = (const float*)d_in[3];
  const float* Wo   = (const float*)d_in[4];
  const float* proj = (const float*)d_in[5];

  size_t off = 0;
  auto alloc = [&](size_t bytes) {
    void* p = (char*)d_ws + off;
    off += (bytes + 255) & ~(size_t)255;
    return p;
  };
  unsigned short* Q   = (unsigned short*)alloc((size_t)BN * CCH * 2);
  unsigned short* Kf  = (unsigned short*)alloc((size_t)BN * CCH * 2);
  unsigned short* V   = (unsigned short*)alloc((size_t)BN * CCH * 2);
  unsigned short* qp  = (unsigned short*)alloc((size_t)BH * NN * MMF * 2);
  unsigned short* kp  = (unsigned short*)alloc((size_t)BH * NN * MMF * 2);
  unsigned short* ctx = (unsigned short*)alloc((size_t)BH * DH * MMF * 2);
  float* ksum   = (float*)alloc((size_t)BH * MMF * 4);
  float* keymax = (float*)alloc((size_t)BH * 4);
  float* denom  = (float*)alloc((size_t)BH * NN * 4);
  float* Obuf   = (float*)alloc((size_t)BN * CCH * 4);

  perf_init_kernel<<<dim3((BH * NN + 255) / 256), dim3(256), 0, stream>>>(ksum, keymax, denom);
  perf_qkv_kernel<<<dim3(BN / 64, CCH / 64, 3), dim3(256), 0, stream>>>(
      x, Wq, Wk, Wv, Q, Kf, V);
  perf_feature_kernel<0><<<dim3(NN / 64, BH), dim3(256), 0, stream>>>(
      Kf, proj, nullptr, nullptr, keymax, nullptr);
  perf_feature_kernel<1><<<dim3(NN / 64, BH), dim3(256), 0, stream>>>(
      Kf, proj, kp, ksum, keymax, nullptr);
  perf_feature_kernel<2><<<dim3(NN / 64, BH), dim3(256), 0, stream>>>(
      Q, proj, qp, ksum, nullptr, denom);
  perf_context_kernel<<<dim3(BH), dim3(256), 0, stream>>>(kp, V, ctx);
  perf_attnout_kernel<<<dim3(NN / 64, BH), dim3(256), 0, stream>>>(qp, ctx, denom, Obuf);
  perf_outproj_kernel<<<dim3(BN / 64, CCH / 64), dim3(256), 0, stream>>>(
      Obuf, Wo, (float*)d_out);
}
